// MentionScore_4286377362075
// MI455X (gfx1250) — matrix operand order repaired and checked by reading, not compiled, not run on hardware
//
#include <hip/hip_runtime.h>
#include <stdint.h>

// ---------------- problem constants ----------------
#define TT    4096
#define AA    1024
#define EE    512
#define NSPAN 32768
#define WMAX  10
#define HID   150
#define FD    20
#define HPAD  160          // HID padded to 10 WMMA N-tiles
#define KG    2592         // concat dim 2580 padded to 81*32
#define NTILES 10
#define WAVES_PER_BLOCK 8
#define ROWS_PER_BLOCK (16 * WAVES_PER_BLOCK)   // 128
#define BSLAB_USHORTS (HPAD * 32)               // 5120 ushorts = 10 KB per k-step

typedef __attribute__((ext_vector_type(16))) __bf16 bf16x16;
typedef __attribute__((ext_vector_type(8)))  float  floatx8;
typedef __attribute__((ext_vector_type(4)))  int    v4i_t;
typedef __attribute__((address_space(1))) v4i_t* as1_v4i_p;
typedef __attribute__((address_space(3))) v4i_t* as3_v4i_p;

union BfVec { bf16x16 v; uint32_t u[8]; uint4 q[2]; };

#if defined(__AMDGCN__) && __has_builtin(__builtin_amdgcn_global_load_async_to_lds_b128)
#define USE_ASYNC 1
#else
#define USE_ASYNC 0
#endif

__device__ __forceinline__ unsigned short f2bf(float f) {
  union { float f; uint32_t u; } x; x.f = f;
  uint32_t r = x.u + 0x7FFFu + ((x.u >> 16) & 1u);   // round-to-nearest-even
  return (unsigned short)(r >> 16);
}

__device__ __forceinline__ floatx8 wmma_bf16(bf16x16 a, bf16x16 b, floatx8 c) {
  return __builtin_amdgcn_wmma_f32_16x16x32_bf16(false, a, false, b, (short)0, c, false, false);
}

__device__ __forceinline__ floatx8 zero8() {
  floatx8 z;
#pragma unroll
  for (int q = 0; q < 8; ++q) z[q] = 0.0f;
  return z;
}

// A-tile 16x32 bf16: lane m=lane&15, half=lane>>4, VGPR j K-pair at:
__device__ __forceinline__ int a_koff(int j, int half) {
  return ((j >> 2) << 4) + ((j & 3) << 1) + half * 8;
}

// 16B global->LDS copy: async on gfx1250, fallback = load + ds_store
__device__ __forceinline__ void copy16_g2l(const unsigned short* g, unsigned short* l) {
#if USE_ASYNC
  __builtin_amdgcn_global_load_async_to_lds_b128((as1_v4i_p)g, (as3_v4i_p)l, 0, 0);
#else
  *(uint4*)l = *(const uint4*)g;
#endif
}

__device__ __forceinline__ void wait_g2l() {
#if USE_ASYNC
#if __has_builtin(__builtin_amdgcn_s_wait_asynccnt)
  __builtin_amdgcn_s_wait_asynccnt(0);
#else
  asm volatile("s_wait_asynccnt 0x0" ::: "memory");
#endif
#endif
}

// cooperative load of one [HPAD x 32] B slab (k0..k0+31) into LDS
__device__ __forceinline__ void load_bslab(const unsigned short* __restrict__ WT, int K, int k0,
                                           unsigned short* __restrict__ smemB) {
#pragma unroll
  for (int it = 0; it < 3; ++it) {                 // 640 chunks / 256 threads
    int c = threadIdx.x + it * 256;
    if (c < (BSLAB_USHORTS / 8)) {
      int n = c >> 2, sub = c & 3;                 // 4x16B chunks per 64B row
      copy16_g2l(WT + (size_t)n * K + k0 + sub * 8, smemB + c * 8);
    }
  }
  wait_g2l();
  __syncthreads();
}

// each lane grabs its 10 B tiles (register layout) from the LDS slab
__device__ __forceinline__ void load_bregs(const unsigned short* __restrict__ smemB,
                                           int nl, int half, BfVec* b) {
#pragma unroll
  for (int nt = 0; nt < NTILES; ++nt) {
    const uint4* p = (const uint4*)(smemB + (nt * 16 + nl) * 32 + half * 8);
    b[nt].q[0] = p[0];
    b[nt].q[1] = p[2];
  }
}

// ---------------- conversion kernels ----------------
__global__ void k_cvt_bf16(const float* __restrict__ src, unsigned short* __restrict__ dst, int n) {
  int i = blockIdx.x * blockDim.x + threadIdx.x;
  if (i < n) dst[i] = f2bf(src[i]);
}

// src [Ksrc x Nsrc] fp32 -> dst [Npad x Kpad] bf16 (transposed, zero-padded)
__global__ void k_transpose_cvt(const float* __restrict__ src, unsigned short* __restrict__ dst,
                                int Ksrc, int Nsrc, int Kpad, int Npad) {
  int i = blockIdx.x * blockDim.x + threadIdx.x;
  if (i >= Kpad * Npad) return;
  int n = i / Kpad, k = i - n * Kpad;
  float v = (k < Ksrc && n < Nsrc) ? src[(size_t)k * Nsrc + n] : 0.0f;
  dst[i] = f2bf(v);
}

// ---------------- GEMM1 (dense A): out = relu(A @ W + b) -> bf16 [M x HPAD] ----------------
__global__ void __launch_bounds__(256)
k_gemm1_dense(const unsigned short* __restrict__ Abf,  // [M x K] bf16 row-major
              const unsigned short* __restrict__ WT,   // [HPAD x K] bf16 (W transposed)
              const float* __restrict__ bias,          // [HID]
              unsigned short* __restrict__ out,        // [M x HPAD]
              int M, int K) {
  __shared__ unsigned short smemB[BSLAB_USHORTS];
  int wave = threadIdx.x >> 5;
  int lane = threadIdx.x & 31;
  int half = lane >> 4;
  int nl   = lane & 15;
  int m0 = (blockIdx.x * WAVES_PER_BLOCK + wave) * 16;

  const unsigned short* rowp = Abf + (size_t)(m0 + nl) * K;
  floatx8 acc[NTILES];
#pragma unroll
  for (int nt = 0; nt < NTILES; ++nt) acc[nt] = zero8();

  for (int k0 = 0; k0 < K; k0 += 32) {
    load_bslab(WT, K, k0, smemB);

    BfVec a;
#pragma unroll
    for (int j = 0; j < 8; ++j)
      a.u[j] = *(const uint32_t*)(rowp + k0 + a_koff(j, half));
    BfVec b[NTILES];
    load_bregs(smemB, nl, half, b);
#pragma unroll
    for (int nt = 0; nt < NTILES; ++nt)
      acc[nt] = wmma_bf16(a.v, b[nt].v, acc[nt]);
    __syncthreads();
  }
#pragma unroll
  for (int nt = 0; nt < NTILES; ++nt) {
    int col = nt * 16 + nl;
    float bv = (col < HID) ? bias[col] : 0.0f;
#pragma unroll
    for (int r = 0; r < 8; ++r) {
      int row = m0 + r + half * 8;
      float v = acc[nt][r] + bv;
      v = v > 0.0f ? v : 0.0f;
      out[(size_t)row * HPAD + col] = f2bf(v);
    }
  }
}

// ---------------- GEMM1 (gathered A) for mention features ----------------
// A row s = [states[start(s)] | states[end(s)] | attn_sum(s) | size_feat(s)] (bf16, K = KG)
__global__ void __launch_bounds__(256)
k_gemm1_mention(const unsigned short* __restrict__ states_b, // [T x A]
                const unsigned short* __restrict__ attnsum,  // [N x 512]
                const unsigned short* __restrict__ sizefeat, // [N x 32]
                const int* __restrict__ starts,
                const int* __restrict__ widths,
                const unsigned short* __restrict__ WT,       // [HPAD x KG]
                const float* __restrict__ bias,
                unsigned short* __restrict__ out,            // [N x HPAD]
                int M) {
  __shared__ unsigned short smemB[BSLAB_USHORTS];
  int wave = threadIdx.x >> 5;
  int lane = threadIdx.x & 31;
  int half = lane >> 4;
  int nl   = lane & 15;
  int m0 = (blockIdx.x * WAVES_PER_BLOCK + wave) * 16;

  int s  = m0 + nl;
  int st = starts[s];
  int en = st + widths[s] - 1;
  const unsigned short* baseS = states_b + (size_t)st * AA;
  const unsigned short* baseE = states_b + (size_t)en * AA;
  const unsigned short* baseA = attnsum  + (size_t)s  * EE;
  const unsigned short* baseF = sizefeat + (size_t)s  * 32;

  floatx8 acc[NTILES];
#pragma unroll
  for (int nt = 0; nt < NTILES; ++nt) acc[nt] = zero8();

  for (int k0 = 0; k0 < KG; k0 += 32) {   // 81 k-steps; each lies in one feature segment
    load_bslab(WT, KG, k0, smemB);

    const unsigned short* rowp;
    int koff;
    if (k0 < 1024)      { rowp = baseS; koff = k0;        }
    else if (k0 < 2048) { rowp = baseE; koff = k0 - 1024; }
    else if (k0 < 2560) { rowp = baseA; koff = k0 - 2048; }
    else                { rowp = baseF; koff = k0 - 2560; }
    BfVec a;
#pragma unroll
    for (int j = 0; j < 8; ++j)
      a.u[j] = *(const uint32_t*)(rowp + koff + a_koff(j, half));
    BfVec b[NTILES];
    load_bregs(smemB, nl, half, b);
#pragma unroll
    for (int nt = 0; nt < NTILES; ++nt)
      acc[nt] = wmma_bf16(a.v, b[nt].v, acc[nt]);
    __syncthreads();
  }
#pragma unroll
  for (int nt = 0; nt < NTILES; ++nt) {
    int col = nt * 16 + nl;
    float bv = (col < HID) ? bias[col] : 0.0f;
#pragma unroll
    for (int r = 0; r < 8; ++r) {
      int row = m0 + r + half * 8;
      float v = acc[nt][r] + bv;
      v = v > 0.0f ? v : 0.0f;
      out[(size_t)row * HPAD + col] = f2bf(v);
    }
  }
}

// ---------------- fused GEMM2 + ReLU + dot(W3) + b3 -> scalar per row ----------------
__global__ void __launch_bounds__(256)
k_gemm2_dot(const unsigned short* __restrict__ Abf, // [M x HPAD]
            const unsigned short* __restrict__ WT,  // [HPAD x HPAD]
            const float* __restrict__ b2,           // [HID]
            const float* __restrict__ w3,           // [HID]
            const float* __restrict__ b3,           // [1]
            float* __restrict__ out, int M) {
  __shared__ unsigned short smemB[BSLAB_USHORTS];
  int wave = threadIdx.x >> 5;
  int lane = threadIdx.x & 31;
  int half = lane >> 4;
  int nl   = lane & 15;
  int m0 = (blockIdx.x * WAVES_PER_BLOCK + wave) * 16;

  const unsigned short* rowp = Abf + (size_t)(m0 + nl) * HPAD;
  floatx8 acc[NTILES];
#pragma unroll
  for (int nt = 0; nt < NTILES; ++nt) acc[nt] = zero8();

  for (int k0 = 0; k0 < HPAD; k0 += 32) {   // 5 steps
    load_bslab(WT, HPAD, k0, smemB);

    BfVec a;
#pragma unroll
    for (int j = 0; j < 8; ++j)
      a.u[j] = *(const uint32_t*)(rowp + k0 + a_koff(j, half));
    BfVec b[NTILES];
    load_bregs(smemB, nl, half, b);
#pragma unroll
    for (int nt = 0; nt < NTILES; ++nt)
      acc[nt] = wmma_bf16(a.v, b[nt].v, acc[nt]);
    __syncthreads();
  }
  // epilogue: relu(acc + b2) dot w3, reduced across the 16 N-lanes of each half
  float part[8];
#pragma unroll
  for (int r = 0; r < 8; ++r) part[r] = 0.0f;
#pragma unroll
  for (int nt = 0; nt < NTILES; ++nt) {
    int col = nt * 16 + nl;
    float bv = (col < HID) ? b2[col] : 0.0f;
    float wv = (col < HID) ? w3[col] : 0.0f;
#pragma unroll
    for (int r = 0; r < 8; ++r) {
      float v = acc[nt][r] + bv;
      v = v > 0.0f ? v : 0.0f;
      part[r] += v * wv;
    }
  }
#pragma unroll
  for (int mask = 1; mask <= 8; mask <<= 1)
#pragma unroll
    for (int r = 0; r < 8; ++r)
      part[r] += __shfl_xor(part[r], mask, 32);
  if (nl == 0) {
    float bb = b3[0];
#pragma unroll
    for (int r = 0; r < 8; ++r)
      out[m0 + r + half * 8] = part[r] + bb;
  }
}

// ---------------- span prep: softmax over <=10 logits, weighted embed sum, width bin ----------------
__global__ void k_span_prep(const float* __restrict__ embeds,     // [T x E]
                            const int* __restrict__ starts,
                            const int* __restrict__ widths,
                            const float* __restrict__ attns,      // [T]
                            const float* __restrict__ width_emb,  // [9 x FD]
                            unsigned short* __restrict__ attnsum, // [N x 512] bf16
                            unsigned short* __restrict__ sizefeat // [N x 32]  bf16
                            ) {
  int warp = (blockIdx.x * blockDim.x + threadIdx.x) >> 5;
  int lane = threadIdx.x & 31;
  if (warp >= NSPAN) return;
  int st = starts[warp], wd = widths[warp];

  float l = (lane < wd) ? attns[st + lane] : -__builtin_inff();
  float mx = l;
#pragma unroll
  for (int off = 16; off > 0; off >>= 1) mx = fmaxf(mx, __shfl_xor(mx, off, 32));
  float e = (lane < wd) ? __expf(l - mx) : 0.0f;
  float sm = e;
#pragma unroll
  for (int off = 16; off > 0; off >>= 1) sm += __shfl_xor(sm, off, 32);
  float wgt = e / sm;

  float acc[16];
#pragma unroll
  for (int t = 0; t < 16; ++t) acc[t] = 0.0f;
  for (int i = 0; i < wd; ++i) {
    float wi = __shfl(wgt, i, 32);
    const float4* rp = (const float4*)(embeds + (size_t)(st + i) * EE) + lane * 4;
#pragma unroll
    for (int q = 0; q < 4; ++q) {
      float4 r = rp[q];
      acc[q * 4 + 0] += wi * r.x;
      acc[q * 4 + 1] += wi * r.y;
      acc[q * 4 + 2] += wi * r.z;
      acc[q * 4 + 3] += wi * r.w;
    }
  }
  unsigned short* op = attnsum + (size_t)warp * EE + lane * 16;
#pragma unroll
  for (int t = 0; t < 16; ++t) op[t] = f2bf(acc[t]);

  const int binsv[8] = {1, 2, 3, 4, 8, 16, 32, 64};
  int bin = 0;
#pragma unroll
  for (int j = 0; j < 8; ++j) bin += (wd >= binsv[j]) ? 1 : 0;
  float sv = (lane < FD) ? width_emb[bin * FD + lane] : 0.0f;
  sizefeat[(size_t)warp * 32 + lane] = f2bf(sv);
}

// ---------------- launcher ----------------
extern "C" void kernel_launch(void* const* d_in, const int* in_sizes, int n_in,
                              void* d_out, int out_size, void* d_ws, size_t ws_size,
                              hipStream_t stream) {
  const float* states = (const float*)d_in[0];
  const float* embeds = (const float*)d_in[1];
  const int*   starts = (const int*)d_in[2];
  const int*   widths = (const int*)d_in[3];
  const float* Wa1 = (const float*)d_in[4];
  const float* ba1 = (const float*)d_in[5];
  const float* Wa2 = (const float*)d_in[6];
  const float* ba2 = (const float*)d_in[7];
  const float* Wa3 = (const float*)d_in[8];
  const float* ba3 = (const float*)d_in[9];
  const float* wemb = (const float*)d_in[10];
  const float* Ws1 = (const float*)d_in[11];
  const float* bs1 = (const float*)d_in[12];
  const float* Ws2 = (const float*)d_in[13];
  const float* bs2 = (const float*)d_in[14];
  const float* Ws3 = (const float*)d_in[15];
  const float* bs3 = (const float*)d_in[16];
  float* out = (float*)d_out;
  (void)in_sizes; (void)n_in; (void)out_size; (void)ws_size;

  char* ws = (char*)d_ws;
  size_t off = 0;
  auto alloc = [&](size_t bytes) -> void* {
    void* p = ws + off;
    off = (off + bytes + 255) & ~(size_t)255;
    return p;
  };
  unsigned short* states_b = (unsigned short*)alloc((size_t)TT * AA * 2);
  unsigned short* Wa1T  = (unsigned short*)alloc((size_t)HPAD * AA * 2);
  unsigned short* Wa2T  = (unsigned short*)alloc((size_t)HPAD * HPAD * 2);
  unsigned short* Ws1T  = (unsigned short*)alloc((size_t)HPAD * KG * 2);
  unsigned short* Ws2T  = (unsigned short*)alloc((size_t)HPAD * HPAD * 2);
  unsigned short* H1    = (unsigned short*)alloc((size_t)TT * HPAD * 2);
  float*          attns = (float*)alloc((size_t)TT * 4);
  unsigned short* asum  = (unsigned short*)alloc((size_t)NSPAN * EE * 2);
  unsigned short* sfeat = (unsigned short*)alloc((size_t)NSPAN * 32 * 2);
  unsigned short* S1    = (unsigned short*)alloc((size_t)NSPAN * HPAD * 2);

  // 1) convert states to bf16
  { int n = TT * AA; k_cvt_bf16<<<(n + 255) / 256, 256, 0, stream>>>(states, states_b, n); }
  // 2) transpose-convert weights to [Npad x Kpad] bf16
  { int n = HPAD * AA;   k_transpose_cvt<<<(n + 255) / 256, 256, 0, stream>>>(Wa1, Wa1T, AA, HID, AA, HPAD); }
  { int n = HPAD * HPAD; k_transpose_cvt<<<(n + 255) / 256, 256, 0, stream>>>(Wa2, Wa2T, HID, HID, HPAD, HPAD); }
  { int n = HPAD * KG;   k_transpose_cvt<<<(n + 255) / 256, 256, 0, stream>>>(Ws1, Ws1T, 2580, HID, KG, HPAD); }
  { int n = HPAD * HPAD; k_transpose_cvt<<<(n + 255) / 256, 256, 0, stream>>>(Ws2, Ws2T, HID, HID, HPAD, HPAD); }

  // 3) attention MLP layer 1: H1 = relu(states @ Wa1 + ba1)   [4096 x 160] bf16
  k_gemm1_dense<<<TT / ROWS_PER_BLOCK, 256, 0, stream>>>(states_b, Wa1T, ba1, H1, TT, AA);
  // 4) attention MLP layers 2+3 fused -> attns [4096] fp32
  k_gemm2_dot<<<TT / ROWS_PER_BLOCK, 256, 0, stream>>>(H1, Wa2T, ba2, Wa3, ba3, attns, TT);
  // 5) per-span softmax + weighted embed sum + width-bin embedding
  k_span_prep<<<(NSPAN * 32) / 256, 256, 0, stream>>>(embeds, starts, widths, attns, wemb, asum, sfeat);
  // 6) mention MLP layer 1 on gathered features: S1 = relu(g @ Ws1 + bs1)  [32768 x 160] bf16
  k_gemm1_mention<<<NSPAN / ROWS_PER_BLOCK, 256, 0, stream>>>(states_b, asum, sfeat, starts, widths,
                                                              Ws1T, bs1, S1, NSPAN);
  // 7) mention MLP layers 2+3 fused -> scores [32768] fp32
  k_gemm2_dot<<<NSPAN / ROWS_PER_BLOCK, 256, 0, stream>>>(S1, Ws2T, bs2, Ws3, bs3, out, NSPAN);
}
